// CausalSelfAttention_24799141167255
// MI455X (gfx1250) — matrix operand order repaired and checked by reading, not compiled, not run on hardware
//
#include <hip/hip_runtime.h>
#include <hip/hip_bf16.h>

#define DM   1024
#define NH   16
#define DH   64
#define BATCH 4
#define SEQ  1024

typedef __attribute__((ext_vector_type(16))) __bf16       v16bf;
typedef __attribute__((ext_vector_type(8)))  float        v8f;
typedef __attribute__((ext_vector_type(4)))  unsigned int u32x4;
typedef __attribute__((ext_vector_type(8)))  unsigned int u32x8;

union FragU { u32x8 u; v16bf h; };

static __device__ inline v16bf make_frag(u32x4 lo, u32x4 hi) {
  FragU t;
  t.u = __builtin_shufflevector(lo, hi, 0, 1, 2, 3, 4, 5, 6, 7);
  return t.h;
}

static __device__ inline unsigned short f2bf(float f) {
  union { float f; unsigned u; } c; c.f = f;
  unsigned u = c.u;
  u += 0x7fffu + ((u >> 16) & 1u);   // round-to-nearest-even
  return (unsigned short)(u >> 16);
}

// 32-bit LDS byte offset of a __shared__ object
static __device__ inline unsigned lds_off(const void* p) {
  return (unsigned)(unsigned long long)(__attribute__((address_space(3))) const char*)p;
}

// CDNA5 async memory->LDS copy, 16B per lane, tracked by ASYNCcnt.
static __device__ inline void async_copy_b128(const void* gsrc, void* ldst) {
  unsigned           l = lds_off(ldst);
  unsigned long long g = (unsigned long long)gsrc;
  asm volatile("global_load_async_to_lds_b128 %0, %1, off"
               :: "v"(l), "v"(g) : "memory");
}

static __device__ inline void wait_asynccnt0() {
  asm volatile("s_wait_asynccnt 0x0" ::: "memory");
}

// CDNA5 Tensor Data Mover: DMA a 2D bf16 tile (tile_d0 x tile_d1, dim0
// contiguous) from global memory into LDS. Descriptor per ISA ch.8;
// groups 2/3 omitted (2D tensor). Issue once per workgroup (wave 0).
static __device__ inline void tdm_load_2d_to_lds(const void* gtile, unsigned ldsaddr,
                                                 unsigned tile_d0, unsigned tile_d1,
                                                 unsigned tensor_d0, unsigned tensor_d1,
                                                 unsigned stride_d0) {
  unsigned long long ga = (unsigned long long)gtile;
  u32x4 g0;
  g0[0] = 1u;                                   // count=1 (valid), load, user mode
  g0[1] = ldsaddr;                              // lds_addr [63:32]
  g0[2] = (unsigned)ga;                         // global_addr [95:64]
  g0[3] = (unsigned)(ga >> 32) | (2u << 30);    // global_addr hi + type=2 (image)
  u32x8 g1;
  g1[0] = 1u << 16;                             // data_size = 1 -> 2 bytes
  g1[1] = (tensor_d0 & 0xffffu) << 16;          // tensor_dim0[15:0] @ [63:48]
  g1[2] = (tensor_d0 >> 16) | ((tensor_d1 & 0xffffu) << 16); // dim0 hi | dim1 lo
  g1[3] = (tensor_d1 >> 16) | (tile_d0 << 16);  // dim1 hi | tile_dim0
  g1[4] = tile_d1;                              // tile_dim1 (tile_dim2 = 0)
  g1[5] = stride_d0;                            // tensor_dim0_stride[31:0]
  g1[6] = 0u;                                   // stride hi | dim1_stride lo
  g1[7] = 0u;                                   // dim1_stride hi (unused for 2D)
  asm volatile("tensor_load_to_lds %0, %1" :: "s"(g0), "s"(g1) : "memory");
}

// ---------------------------------------------------------------- fp32 -> bf16
__global__ void cvt_bf16_kernel(const float* __restrict__ in,
                                unsigned short* __restrict__ out, int n4) {
  int i = blockIdx.x * blockDim.x + threadIdx.x;
  int stride = gridDim.x * blockDim.x;
  for (; i < n4; i += stride) {
    float4 v = ((const float4*)in)[i];
    ushort4 o;
    o.x = f2bf(v.x); o.y = f2bf(v.y); o.z = f2bf(v.z); o.w = f2bf(v.w);
    ((ushort4*)out)[i] = o;
  }
}

// ---------------------------------------------------------------- WMMA GEMM
// Y[M,N] = A[M,K] * W[N,K]^T  (A,W bf16 row-major, K = 1024)
// Block tile 128x128, 8 waves, wave tile 32x64 (2x4 WMMA 16x16 tiles).
// Double-buffered LDS staged with async copies: tile k+1 overlaps WMMA on k.
// MODE 0: scatter bf16 into Q/K/V [B,H,T,64].  MODE 1: f32 row-major out.
template <int MODE>
__global__ __launch_bounds__(256) void gemm_bf16_kernel(
    const unsigned short* __restrict__ A,
    const unsigned short* __restrict__ W,
    unsigned short* __restrict__ qo,
    unsigned short* __restrict__ ko,
    unsigned short* __restrict__ vo,
    float* __restrict__ outf) {
  __shared__ __align__(16) unsigned short sA[2][128 * 32];
  __shared__ __align__(16) unsigned short sB[2][128 * 32];
  const int K = 1024;
  const int KT = K / 32;
  const int tid = threadIdx.x;
  const int lane = tid & 31, wid = tid >> 5;
  const int ln = lane & 15, hi = lane >> 4;
  const int wm = wid >> 1, wn = wid & 1;
  const int mblk = blockIdx.y * 128;
  const int nblk = blockIdx.x * 128;

  v8f acc[2][4] = {};

  auto stage = [&](int buf, int k0) {
#pragma unroll
    for (int j = 0; j < 2; ++j) {
      int c = tid + j * 256;
      int row = c >> 2, off = (c & 3) * 8;
      async_copy_b128(&A[(size_t)(mblk + row) * K + k0 + off],
                      &sA[buf][row * 32 + off]);
      async_copy_b128(&W[(size_t)(nblk + row) * K + k0 + off],
                      &sB[buf][row * 32 + off]);
    }
  };

  stage(0, 0);

  for (int kt = 0; kt < KT; ++kt) {
    const int cur = kt & 1;
    wait_asynccnt0();
    __syncthreads();               // buffer `cur` visible to all waves
    if (kt + 1 < KT) stage(cur ^ 1, (kt + 1) * 32);  // prefetch next tile

    v16bf af[2];
#pragma unroll
    for (int mt = 0; mt < 2; ++mt) {
      int r = wm * 32 + mt * 16 + ln;
      u32x4 lo = *(const u32x4*)&sA[cur][r * 32 + 8 * hi];
      u32x4 hq = *(const u32x4*)&sA[cur][r * 32 + 16 + 8 * hi];
      af[mt] = make_frag(lo, hq);
    }
    v16bf bfr[4];
#pragma unroll
    for (int nt = 0; nt < 4; ++nt) {
      int r = wn * 64 + nt * 16 + ln;
      u32x4 lo = *(const u32x4*)&sB[cur][r * 32 + 8 * hi];
      u32x4 hq = *(const u32x4*)&sB[cur][r * 32 + 16 + 8 * hi];
      bfr[nt] = make_frag(lo, hq);
    }
#pragma unroll
    for (int mt = 0; mt < 2; ++mt)
#pragma unroll
      for (int nt = 0; nt < 4; ++nt)
        acc[mt][nt] = __builtin_amdgcn_wmma_f32_16x16x32_bf16(
            false, af[mt], false, bfr[nt], (short)0, acc[mt][nt], false, false);
    __syncthreads();               // all reads of `cur` done before re-staging
  }

#pragma unroll
  for (int mt = 0; mt < 2; ++mt) {
#pragma unroll
    for (int nt = 0; nt < 4; ++nt) {
#pragma unroll
      for (int r = 0; r < 8; ++r) {
        int m = mblk + wm * 32 + mt * 16 + r + 8 * hi;
        int n = nblk + wn * 64 + nt * 16 + ln;
        float v = acc[mt][nt][r];
        if (MODE == 0) {
          int sel = n >> 10, nr = n & 1023;
          int h = nr >> 6, d = nr & 63;
          int b = m >> 10, t = m & 1023;
          size_t idx = (((size_t)(b * NH + h)) * SEQ + t) * DH + d;
          unsigned short bv = f2bf(v);
          if (sel == 0)      qo[idx] = bv;
          else if (sel == 1) ko[idx] = bv;
          else               vo[idx] = bv;
        } else {
          outf[(size_t)m * DM + n] = v;
        }
      }
    }
  }
}

// ---------------------------------------------------------------- flash attention
// One block (4 waves) handles 64 query rows of one (b,h); wave owns 16 rows.
// K tile staged by the Tensor Data Mover (wave 0 issues descriptor).
__global__ __launch_bounds__(128) void flash_attn_kernel(
    const unsigned short* __restrict__ Q,   // [B*H, T, 64] bf16
    const unsigned short* __restrict__ Km,
    const unsigned short* __restrict__ Vm,
    unsigned short* __restrict__ Y) {       // [B, T, C] bf16
  __shared__ __align__(16) unsigned short sK[32 * 64];
  __shared__ __align__(16) unsigned short sVt[64 * 32];
  __shared__ __align__(16) unsigned short sP[4 * 16 * 32];

  const int tid = threadIdx.x;
  const int lane = tid & 31, wid = tid >> 5;
  const int ln = lane & 15, hi = lane >> 4;

  const int qt = blockIdx.x & 15;
  const int bh = blockIdx.x >> 4;
  const int b = bh >> 4, h = bh & 15;

  const unsigned short* Qh = Q + (size_t)bh * SEQ * DH;
  const unsigned short* Kh = Km + (size_t)bh * SEQ * DH;
  const unsigned short* Vh = Vm + (size_t)bh * SEQ * DH;

  const int qrow0 = qt * 64 + wid * 16;

  // Q fragments for both 32-deep k-steps over Dh, kept in registers.
  v16bf aq[2];
#pragma unroll
  for (int s = 0; s < 2; ++s) {
    const unsigned short* base = Qh + (size_t)(qrow0 + ln) * DH + 32 * s;
    u32x4 lo = *(const u32x4*)(base + 8 * hi);
    u32x4 hq = *(const u32x4*)(base + 16 + 8 * hi);
    aq[s] = make_frag(lo, hq);
  }

  v8f o[4] = {};
  float mrun[8], lrun[8];
#pragma unroll
  for (int r = 0; r < 8; ++r) { mrun[r] = -__builtin_inff(); lrun[r] = 0.f; }

  const unsigned sK_lds = lds_off(&sK[0]);

  const int ntiles = (qt + 1) * 2;
  for (int kt = 0; kt < ntiles; ++kt) {
    const int kbase = kt * 32;
    // K tile [32 keys][64 d] via TDM: one descriptor from wave 0
    if (wid == 0) {
      tdm_load_2d_to_lds(Kh + (size_t)kbase * DH, sK_lds,
                         /*tile d0,d1*/ DH, 32,
                         /*tensor d0,d1*/ DH, SEQ,
                         /*dim0 stride*/ DH);
    }
    // stage V tile transposed: sVt[d][key] (needs lane shuffling -> manual)
#pragma unroll
    for (int j = 0; j < 2; ++j) {
      int c = tid + j * 128;
      int kr = c >> 3, off = (c & 7) * 8;
      u32x4 g = *(const u32x4*)&Vh[(size_t)(kbase + kr) * DH + off];
#pragma unroll
      for (int e = 0; e < 4; ++e) {
        unsigned w = g[e];
        sVt[(off + 2 * e) * 32 + kr]     = (unsigned short)(w & 0xffffu);
        sVt[(off + 2 * e + 1) * 32 + kr] = (unsigned short)(w >> 16);
      }
    }
    if (wid == 0) __builtin_amdgcn_s_wait_tensorcnt(0);
    __syncthreads();

    // S = Q * K^T  (two 16-wide key tiles, accumulate over Dh)
    v8f s0 = {}, s1 = {};
#pragma unroll
    for (int s = 0; s < 2; ++s) {
      u32x4 lo0 = *(const u32x4*)&sK[ln * 64 + 32 * s + 8 * hi];
      u32x4 hq0 = *(const u32x4*)&sK[ln * 64 + 32 * s + 16 + 8 * hi];
      v16bf bk0 = make_frag(lo0, hq0);
      u32x4 lo1 = *(const u32x4*)&sK[(16 + ln) * 64 + 32 * s + 8 * hi];
      u32x4 hq1 = *(const u32x4*)&sK[(16 + ln) * 64 + 32 * s + 16 + 8 * hi];
      v16bf bk1 = make_frag(lo1, hq1);
      s0 = __builtin_amdgcn_wmma_f32_16x16x32_bf16(false, aq[s], false, bk0,
                                                   (short)0, s0, false, false);
      s1 = __builtin_amdgcn_wmma_f32_16x16x32_bf16(false, aq[s], false, bk1,
                                                   (short)0, s1, false, false);
    }
    s0 *= 0.125f;  // 1/sqrt(64)
    s1 *= 0.125f;

    // online softmax (rows live across 16 lanes of each half-wave)
    const int wb = wid * 16 * 32;
#pragma unroll
    for (int r = 0; r < 8; ++r) {
      const int qrow = qrow0 + r + 8 * hi;
      float v0 = (kbase + ln      <= qrow) ? s0[r] : -__builtin_inff();
      float v1 = (kbase + 16 + ln <= qrow) ? s1[r] : -__builtin_inff();
      float mc = fmaxf(v0, v1);
#pragma unroll
      for (int off = 8; off >= 1; off >>= 1)
        mc = fmaxf(mc, __shfl_xor(mc, off, 32));
      float mnew  = fmaxf(mrun[r], mc);
      float alpha = __expf(mrun[r] - mnew);
      float p0 = __expf(v0 - mnew);
      float p1 = __expf(v1 - mnew);
      float ps = p0 + p1;
#pragma unroll
      for (int off = 8; off >= 1; off >>= 1)
        ps += __shfl_xor(ps, off, 32);
      lrun[r] = lrun[r] * alpha + ps;
      mrun[r] = mnew;
#pragma unroll
      for (int nt = 0; nt < 4; ++nt) o[nt][r] *= alpha;
      sP[wb + (r + 8 * hi) * 32 + ln]      = f2bf(p0);
      sP[wb + (r + 8 * hi) * 32 + 16 + ln] = f2bf(p1);
    }
    // per-wave P buffer: same-wave DS ops are in-order; fence the compiler + DScnt
    asm volatile("s_wait_dscnt 0x0" ::: "memory");

    // P as A-fragment
    u32x4 plo = *(const u32x4*)&sP[wb + ln * 32 + 8 * hi];
    u32x4 phq = *(const u32x4*)&sP[wb + ln * 32 + 16 + 8 * hi];
    v16bf ap = make_frag(plo, phq);

    // O += P * V
#pragma unroll
    for (int nt = 0; nt < 4; ++nt) {
      int d = nt * 16 + ln;
      u32x4 lo = *(const u32x4*)&sVt[d * 32 + 8 * hi];
      u32x4 hq = *(const u32x4*)&sVt[d * 32 + 16 + 8 * hi];
      v16bf bv = make_frag(lo, hq);
      o[nt] = __builtin_amdgcn_wmma_f32_16x16x32_bf16(false, ap, false, bv,
                                                      (short)0, o[nt], false, false);
    }
    __syncthreads();
  }

  // normalize and write Y[b, t, h*64 + d]
#pragma unroll
  for (int nt = 0; nt < 4; ++nt) {
#pragma unroll
    for (int r = 0; r < 8; ++r) {
      int t = qrow0 + r + 8 * hi;
      int dcol = h * DH + nt * 16 + ln;
      float v = o[nt][r] / lrun[r];
      Y[((size_t)b * SEQ + t) * DM + dcol] = f2bf(v);
    }
  }
}

// ---------------------------------------------------------------- launcher
extern "C" void kernel_launch(void* const* d_in, const int* in_sizes, int n_in,
                              void* d_out, int out_size, void* d_ws, size_t ws_size,
                              hipStream_t stream) {
  const float* x     = (const float*)d_in[0];
  const float* w_qkv = (const float*)d_in[1];
  const float* w_out = (const float*)d_in[2];
  float* out = (float*)d_out;

  char* w = (char*)d_ws;
  const size_t MB = 1024 * 1024;
  unsigned short* xb  = (unsigned short*)(w + 0);        // 8 MiB  x bf16
  unsigned short* wqb = (unsigned short*)(w + 8 * MB);   // 6 MiB  w_qkv bf16
  unsigned short* wob = (unsigned short*)(w + 14 * MB);  // 2 MiB  w_out bf16
  unsigned short* qb  = (unsigned short*)(w + 16 * MB);  // 8 MiB  Q [B,H,T,64]
  unsigned short* kb  = (unsigned short*)(w + 24 * MB);  // 8 MiB  K
  unsigned short* vb  = (unsigned short*)(w + 32 * MB);  // 8 MiB  V
  unsigned short* yab = (unsigned short*)(w + 40 * MB);  // 8 MiB  attn out [B,T,C]

  cvt_bf16_kernel<<<512, 256, 0, stream>>>(x,     xb,  (BATCH * SEQ * DM) / 4);
  cvt_bf16_kernel<<<512, 256, 0, stream>>>(w_qkv, wqb, (3 * DM * DM) / 4);
  cvt_bf16_kernel<<<256, 256, 0, stream>>>(w_out, wob, (DM * DM) / 4);

  dim3 g1(3 * DM / 128, BATCH * SEQ / 128);
  gemm_bf16_kernel<0><<<g1, 256, 0, stream>>>(xb, wqb, qb, kb, vb, nullptr);

  flash_attn_kernel<<<BATCH * NH * (SEQ / 64), 128, 0, stream>>>(qb, kb, vb, yab);

  dim3 g2(DM / 128, BATCH * SEQ / 128);
  gemm_bf16_kernel<1><<<g2, 256, 0, stream>>>(yab, wob, nullptr, nullptr, nullptr, out);
}
